// EqEmbedding_72808285602177
// MI455X (gfx1250) — matrix operand order repaired and checked by reading, not compiled, not run on hardware
//
#include <hip/hip_runtime.h>
#include <hip/hip_bf16.h>
#include <math.h>

// ---------------------------------------------------------------------------
// EqEmbedding for MI455X (gfx1250, wave32, WMMA).
//
// g(1x112) = rbf(1x16) @ M(16x112), M folded from W_rbf/x_s/Wp0/Wp1/Wp2.
//   cols  0..63  -> node ch c                      (e0, sh0 == 1)
//   cols 64..95  -> node ch 64 + d*3 + m, x sh1[m] (d = col-64)
//   cols 96..111 -> node ch 160 + d*5 + m, x sh2[m](d = col-96)
//
// WMMA: v_wmma_f32_16x16x32_f16, K=16 zero-padded to 32.
// Per wave: 32 edges = two 16-row A tiles x 7 column tiles = 14 WMMAs.
// Scatter: branch-free, 32-bit voffsets, per-tile constants folded into the
// global atomic's immediate offset.
// ---------------------------------------------------------------------------

typedef __attribute__((ext_vector_type(16))) _Float16 v16h;
typedef __attribute__((ext_vector_type(8)))  float    v8f;

#define TMUL0 64
#define TNODE_DIM 240
#define TNBASIS 16
#define TNCOLS 112

__device__ __forceinline__ void atomAddF(float* p, float v) {
    unsafeAtomicAdd(p, v);   // hardware global_atomic_add_f32
}

// ---------------------------------------------------------------------------
// Kernel 1: combined matrix M (16 x 112) as f16, column-major with 32 rows
// (rows 16..31 zero) so a WMMA B fragment is one contiguous 32B load/lane.
// ---------------------------------------------------------------------------
__global__ void precompute_M(const float* __restrict__ w_expand,
                             const float* __restrict__ b_expand,
                             const float* __restrict__ W_rbf,   // (16, 192)
                             const float* __restrict__ Wp0,     // (64, 64)
                             const float* __restrict__ Wp1,     // (64, 32)
                             const float* __restrict__ Wp2,     // (64, 16)
                             _Float16* __restrict__ Mh)         // (112 x 32)
{
    int idx = blockIdx.x * blockDim.x + threadIdx.x;
    if (idx >= TNCOLS * 32) return;
    int c = idx >> 5;
    int k = idx & 31;
    float v = 0.0f;
    if (k < TNBASIS) {
        const float* Wp;
        int pbase, ncol, j;
        if (c < 64)      { Wp = Wp0; pbase = 0;   ncol = 64; j = c;      }
        else if (c < 96) { Wp = Wp1; pbase = 64;  ncol = 32; j = c - 64; }
        else             { Wp = Wp2; pbase = 128; ncol = 16; j = c - 96; }
        float acc = 0.0f;
        for (int cc = 0; cc < TMUL0; ++cc) {
            float xs = w_expand[cc] + b_expand[cc];
            acc = fmaf(W_rbf[k * 192 + pbase + cc] * xs, Wp[cc * ncol + j], acc);
        }
        v = acc * 0.125f;      // 1/sqrt(64)
    }
    Mh[c * 32 + k] = (_Float16)v;
}

// ---------------------------------------------------------------------------
// Kernel 2: node_emb[:, :64] = W_atom[at_no]/sqrt(119) + b_atom ; rest 0.
// ---------------------------------------------------------------------------
__global__ void init_nodes(const int*   __restrict__ at_no,
                           const float* __restrict__ W_atom,
                           const float* __restrict__ b_atom,
                           float* __restrict__ node_emb, int N)
{
    long long idx = (long long)blockIdx.x * blockDim.x + threadIdx.x;
    long long total = (long long)N * TNODE_DIM;
    if (idx >= total) return;
    int i = (int)(idx / TNODE_DIM);
    int c = (int)(idx % TNODE_DIM);
    float v = 0.0f;
    if (c < TMUL0)
        v = fmaf(W_atom[at_no[i] * TMUL0 + c], 0.09166984970282113f, b_atom[c]);
    node_emb[idx] = v;
}

// ---------------------------------------------------------------------------
// Kernel 3: per-edge kernel. 256 threads = 8 waves, 32 edges per wave.
// ---------------------------------------------------------------------------
__global__ __launch_bounds__(256) void eq_edge_kernel(
    const float* __restrict__ pos,
    const int*   __restrict__ edge_index,   // (2, E)
    const _Float16* __restrict__ Mh,        // (112 x 32) f16
    float* __restrict__ node_emb,
    float* __restrict__ rbf_out,            // (E, 16)
    float* __restrict__ rsh_out,            // (E, 9)
    int E)
{
    __shared__ _Float16 s_rbf[8][32][16];
    __shared__ int      s_dstoff[8][32];    // dst * 240 (0 if inactive)
    __shared__ float    s_sh1[8][32][3];
    __shared__ float    s_sh2[8][32][5];

    const int tid  = threadIdx.x;
    const int lane = tid & 31;
    const int wave = tid >> 5;
    const int e    = blockIdx.x * 256 + tid;
    const bool active = (e < E);
    const int  ec  = active ? e : (E - 1);

    const int src = edge_index[ec];
    const int dst = edge_index[E + ec];

    // pos permuted [1,2,0]: new coords = (y, z, x)
    float svx = pos[src * 3 + 1], svy = pos[src * 3 + 2], svz = pos[src * 3 + 0];
    float dvx = pos[dst * 3 + 1], dvy = pos[dst * 3 + 2], dvz = pos[dst * 3 + 0];
    float vx = dvx - svx, vy = dvy - svy, vz = dvz - svz;
    float dist = sqrtf(fmaf(vx, vx, fmaf(vy, vy, vz * vz)));
    float d = fmaxf(dist, 1e-8f);

    // cutoff polynomial fc = 1 - 21 x^5 + 35 x^6 - 15 x^7, x = d/5
    float xr = d * 0.2f;
    float x5 = (xr * xr) * (xr * xr) * xr;
    float fc = fmaf(x5, fmaf(xr, fmaf(-15.0f, xr, 35.0f), -21.0f), 1.0f);
    fc = (d < 5.0f) ? fc : 0.0f;
    float pref = 0.6324555320336759f / d * fc;   // sqrt(2/5)/d * fc
    // inactive edges contribute exact zeros through the whole scatter path
    pref = active ? pref : 0.0f;

    // sin(n*theta) via Chebyshev recurrence
    float s1, c1;
    sincosf(d * 0.6283185307179586f, &s1, &c1);  // theta = pi*d/5
    float twoc = 2.0f * c1;
    float skm1 = 0.0f, sk = s1;
    float rb[TNBASIS];
#pragma unroll
    for (int n = 0; n < TNBASIS; ++n) {
        float r = pref * sk;
        rb[n] = r;
        s_rbf[wave][lane][n] = (_Float16)r;
        float nx = fmaf(twoc, sk, -skm1);
        skm1 = sk; sk = nx;
    }
    if (active) {
        float4* ro = (float4*)(rbf_out + (size_t)e * TNBASIS);
#pragma unroll
        for (int q = 0; q < 4; ++q)
            ro[q] = make_float4(rb[4*q], rb[4*q+1], rb[4*q+2], rb[4*q+3]);
    }

    // real spherical harmonics of u = vec/d
    float inv_d = 1.0f / d;
    float ux = vx * inv_d, uy = vy * inv_d, uz = vz * inv_d;
    const float s3 = 1.7320508075688772f, s5 = 2.23606797749979f, s15 = 3.872983346207417f;
    float sh[9];
    sh[0] = 1.0f;
    sh[1] = s3 * ux; sh[2] = s3 * uy; sh[3] = s3 * uz;
    sh[4] = s15 * ux * uz;
    sh[5] = s15 * ux * uy;
    sh[6] = s5 * (uy * uy - 0.5f * (ux * ux + uz * uz));
    sh[7] = s15 * uy * uz;
    sh[8] = 0.5f * s15 * (uz * uz - ux * ux);
    if (active) {
#pragma unroll
        for (int j = 0; j < 9; ++j) rsh_out[(size_t)e * 9 + j] = sh[j];
    }
    s_dstoff[wave][lane] = (active ? dst : 0) * TNODE_DIM;
    s_sh1[wave][lane][0] = sh[1];
    s_sh1[wave][lane][1] = sh[2];
    s_sh1[wave][lane][2] = sh[3];
#pragma unroll
    for (int m = 0; m < 5; ++m) s_sh2[wave][lane][m] = sh[4 + m];

    __syncthreads();

    // --- A fragments (contiguous k = 16*hs + i; hs=1 half is zero padding) ---
    const int row = lane & 15;
    const int hs  = lane >> 4;
    v16h a0 = {}; v16h a1 = {};
    const _Float16* pa0 = &s_rbf[wave][row][0];
    const _Float16* pa1 = &s_rbf[wave][16 + row][0];
    if (hs == 0) {
#pragma unroll
    for (int i = 0; i < 16; ++i) { a0[i] = pa0[i]; a1[i] = pa1[i]; }
    }

    // --- hoist the 16 destination offsets this lane will scatter to ---
    // c0 fragment rows -> edge slots hs*8 + r ; c1 -> 16 + hs*8 + r
    unsigned d0[8], d1[8];
#pragma unroll
    for (int r = 0; r < 8; ++r) {
        d0[r] = (unsigned)s_dstoff[wave][hs * 8 + r];
        d1[r] = (unsigned)s_dstoff[wave][16 + hs * 8 + r];
    }

    auto loadB = [&](int t) {
        const _Float16* pb = Mh + ((t * 16 + row) * 32 + hs * 16);
        v16h b;
#pragma unroll
        for (int i = 0; i < 16; ++i) b[i] = pb[i];
        return b;
    };
    auto mma = [&](v16h a, v16h b) {
        v8f c = {};
        return __builtin_amdgcn_wmma_f32_16x16x32_f16(false, a, false, b,
                                                      (short)0, c, false, false);
    };

    // ---- Phase A: tiles 0..3, node channel = t*16 + row (folds to imm) ----
    {
        unsigned o0[8], o1[8];
#pragma unroll
        for (int r = 0; r < 8; ++r) { o0[r] = d0[r] + row; o1[r] = d1[r] + row; }
#pragma unroll
        for (int t = 0; t < 4; ++t) {
            v16h b = loadB(t);
            v8f c0 = mma(a0, b);
            v8f c1 = mma(a1, b);
#pragma unroll
            for (int r = 0; r < 8; ++r) {
                atomAddF(node_emb + o0[r] + t * 16, c0[r]);
                atomAddF(node_emb + o1[r] + t * 16, c1[r]);
            }
        }
    }

    // ---- Phase B: tiles 4,5 -> g1; node ch = 64 + (t-4)*48 + row*3 + m ----
    {
        const unsigned row3 = (unsigned)(row * 3);
        unsigned o0[8], o1[8];
#pragma unroll
        for (int r = 0; r < 8; ++r) { o0[r] = d0[r] + row3; o1[r] = d1[r] + row3; }
#pragma unroll
        for (int t = 4; t < 6; ++t) {
            v16h b = loadB(t);
            v8f c0 = mma(a0, b);
            v8f c1 = mma(a1, b);
            const int cb = 64 + (t - 4) * 48;   // compile-time after unroll
#pragma unroll
            for (int r = 0; r < 8; ++r) {
                const float* p1 = s_sh1[wave][hs * 8 + r];
                float h0 = p1[0], h1 = p1[1], h2 = p1[2];
                atomAddF(node_emb + o0[r] + (cb + 0), c0[r] * h0);
                atomAddF(node_emb + o0[r] + (cb + 1), c0[r] * h1);
                atomAddF(node_emb + o0[r] + (cb + 2), c0[r] * h2);
                const float* q1 = s_sh1[wave][16 + hs * 8 + r];
                float g0 = q1[0], g1 = q1[1], g2 = q1[2];
                atomAddF(node_emb + o1[r] + (cb + 0), c1[r] * g0);
                atomAddF(node_emb + o1[r] + (cb + 1), c1[r] * g1);
                atomAddF(node_emb + o1[r] + (cb + 2), c1[r] * g2);
            }
        }
    }

    // ---- Phase C: tile 6 -> g2; node ch = 160 + row*5 + m ----
    {
        const unsigned row5 = (unsigned)(row * 5);
        unsigned o0[8], o1[8];
#pragma unroll
        for (int r = 0; r < 8; ++r) { o0[r] = d0[r] + row5; o1[r] = d1[r] + row5; }
        v16h b = loadB(6);
        v8f c0 = mma(a0, b);
        v8f c1 = mma(a1, b);
#pragma unroll
        for (int r = 0; r < 8; ++r) {
            const float* p2 = s_sh2[wave][hs * 8 + r];
#pragma unroll
            for (int m = 0; m < 5; ++m)
                atomAddF(node_emb + o0[r] + (160 + m), c0[r] * p2[m]);
            const float* q2 = s_sh2[wave][16 + hs * 8 + r];
#pragma unroll
            for (int m = 0; m < 5; ++m)
                atomAddF(node_emb + o1[r] + (160 + m), c1[r] * q2[m]);
        }
    }
}

// ---------------------------------------------------------------------------
extern "C" void kernel_launch(void* const* d_in, const int* in_sizes, int n_in,
                              void* d_out, int out_size, void* d_ws, size_t ws_size,
                              hipStream_t stream) {
    const int*   at_no      = (const int*)  d_in[0];
    const float* pos        = (const float*)d_in[1];
    const int*   edge_index = (const int*)  d_in[2];
    const float* W_atom     = (const float*)d_in[3];
    const float* b_atom     = (const float*)d_in[4];
    const float* w_expand   = (const float*)d_in[5];
    const float* b_expand   = (const float*)d_in[6];
    const float* W_rbf      = (const float*)d_in[7];
    const float* Wp0        = (const float*)d_in[8];
    const float* Wp1        = (const float*)d_in[9];
    const float* Wp2        = (const float*)d_in[10];

    const int N = in_sizes[0];
    const int E = in_sizes[2] / 2;

    float* node_emb = (float*)d_out;                        // N * 240
    float* rbf_out  = node_emb + (size_t)N * TNODE_DIM;     // E * 16
    float* rsh_out  = rbf_out  + (size_t)E * TNBASIS;       // E * 9

    _Float16* Mh = (_Float16*)d_ws;                         // 112*32 f16 = 7168 B

    precompute_M<<<(TNCOLS * 32 + 255) / 256, 256, 0, stream>>>(
        w_expand, b_expand, W_rbf, Wp0, Wp1, Wp2, Mh);

    long long node_elems = (long long)N * TNODE_DIM;
    init_nodes<<<(int)((node_elems + 255) / 256), 256, 0, stream>>>(
        at_no, W_atom, b_atom, node_emb, N);

    eq_edge_kernel<<<(E + 255) / 256, 256, 0, stream>>>(
        pos, edge_index, Mh, node_emb, rbf_out, rsh_out, E);
}
